// xDeepFM_21139829031302
// MI455X (gfx1250) — compile-verified, matrix operand order checked by reading, not run on hardware
//
#include <hip/hip_runtime.h>
#include <hip/hip_bf16.h>
#include <cstdint>

// ---------------------------------------------------------------------------
// xDeepFM forward for MI455X (gfx1250, wave32, WMMA 16x16x32 f16->f32)
//   B=8192, NUM=13, NCAT=26, V=100000, D=16, M=39 (padded to 40)
//   CIN0: GEMM [131072, 1521->1536, 128]  A built on the fly from x0 (x0*x0)
//   CIN1: GEMM [131072, 2496, 128]        A built on the fly (hid*x0)
//   DNN : GEMMs [8192,640,256] [8192,256,128] [8192,128,64] + batch-BN + relu
// Tile staging uses gfx1250 GLOBAL_LOAD_ASYNC_TO_LDS_B128 (ASYNCcnt).
// ---------------------------------------------------------------------------

typedef __attribute__((ext_vector_type(16))) _Float16 v16h;
typedef __attribute__((ext_vector_type(8)))  _Float16 v8h;
typedef __attribute__((ext_vector_type(8)))  float    v8f;

#define B_SZ  8192
#define NUMF  13
#define NCATF 26
#define VOC   100000
#define DD    16
#define MM    39
#define MP    40   // padded field count (index 39 is a zero column)

union HU16 { v16h v; _Float16 h[16]; };

static __device__ __forceinline__ v16h cat8(v8h lo, v8h hi) {
  return __builtin_shufflevector(lo, hi, 0,1,2,3,4,5,6,7,8,9,10,11,12,13,14,15);
}

static __device__ __forceinline__ v8f wmma_f16(v16h a, v16h b, v8f c) {
  // v_wmma_f32_16x16x32_f16  (neg_a, A, neg_b, B, c_mod, C, reuse_a, reuse_b)
  return __builtin_amdgcn_wmma_f32_16x16x32_f16(false, a, false, b, (short)0, c,
                                                false, false);
}

// gfx1250 async copy: 16 bytes global -> LDS per active lane (ASYNCcnt).
static __device__ __forceinline__ void async_g2l_b128(unsigned lds_off,
                                                      const void* gptr) {
  unsigned long long ga = (unsigned long long)(uintptr_t)gptr;
  asm volatile("global_load_async_to_lds_b128 %0, %1, off"
               :
               : "v"(lds_off), "v"(ga)
               : "memory");
}
static __device__ __forceinline__ void wait_asynccnt0() {
  asm volatile("s_wait_asynccnt 0x0" ::: "memory");
}

// ---------------------------------------------------------------------------
// Utility kernels
// ---------------------------------------------------------------------------
__global__ void zero_kernel(float* __restrict__ p, int n) {
  int i = blockIdx.x * blockDim.x + threadIdx.x;
  if (i < n) p[i] = 0.0f;
}

// linear part: scalar = sum(lin_table) + lin_bias
__global__ void lin_kernel(const float* __restrict__ lin_table,
                           const float* __restrict__ lin_bias,
                           float* __restrict__ out) {
  __shared__ float s[64];
  int t = threadIdx.x;
  s[t] = (t < MM) ? lin_table[t] : 0.0f;
  __syncthreads();
  for (int o = 32; o > 0; o >>= 1) { if (t < o) s[t] += s[t + o]; __syncthreads(); }
  if (t == 0) *out = s[0] + lin_bias[0];
}

// ---------------------------------------------------------------------------
// Build x0 transposed [b][d][m] as f16 with rowlen MP (zero pad at m=39).
// numeric: x0[b,i,d] = num[b,i]*num_w[i,d]+num_b[i,d]; cat: embedding gather.
// ---------------------------------------------------------------------------
__global__ void build_x0_kernel(const float* __restrict__ num,
                                const int*   __restrict__ cat_idx,
                                const float* __restrict__ num_w,
                                const float* __restrict__ num_b,
                                const float* __restrict__ cat_emb,
                                _Float16*    __restrict__ x0t) {
  int t = blockIdx.x * blockDim.x + threadIdx.x;
  if (t >= B_SZ * MM) return;
  int b = t / MM, m = t - b * MM;
  float vals[DD];
  if (m < NUMF) {
    float x = num[b * NUMF + m];
#pragma unroll
    for (int d = 0; d < DD; ++d) vals[d] = x * num_w[m * DD + d] + num_b[m * DD + d];
  } else {
    int c = m - NUMF;
    int idx = cat_idx[b * NCATF + c];
    const float* e = cat_emb + ((size_t)c * VOC + (size_t)idx) * DD;
#pragma unroll
    for (int d = 0; d < DD; ++d) vals[d] = e[d];
  }
#pragma unroll
  for (int d = 0; d < DD; ++d)
    x0t[((size_t)b * DD + d) * MP + m] = (_Float16)vals[d];
  if (m == 0) {  // zero pad column
#pragma unroll
    for (int d = 0; d < DD; ++d)
      x0t[((size_t)b * DD + d) * MP + MM] = (_Float16)0.0f;
  }
}

// ---------------------------------------------------------------------------
// Pre-swizzle weights into WMMA B-fragment-major layout:
//   frag[(kt*NT+nt)*32*16 + lane*16 + j] = B[k, n]  where
//   n = nt*16 + (lane&15),  k = kt*32 + khi + (j<8 ? j : j+8), khi=(lane>=16)?8:0
// modes: 0 cin_w0[128,1521]; 1 cin_w1[128,2496];
//        2 dnn w0[624,256] with k=d*40+m -> src row m*16+d;
//        3 dnn w1[256,128]; 4 dnn w2[128,64]
// ---------------------------------------------------------------------------
__global__ void prep_bfrag_kernel(const float* __restrict__ W,
                                  _Float16* __restrict__ frag,
                                  int mode, int NT, int total) {
  int gid = blockIdx.x * blockDim.x + threadIdx.x;
  if (gid >= total) return;
  int j    = gid & 15;
  int lane = (gid >> 4) & 31;
  int q    = gid >> 9;
  int nt   = q % NT;
  int kt   = q / NT;
  int khi  = (lane >= 16) ? 8 : 0;
  int kl   = khi + ((j < 8) ? j : (j + 8));
  int k    = kt * 32 + kl;
  int n    = nt * 16 + (lane & 15);
  float v = 0.0f;
  switch (mode) {
    case 0: if (k < 1521) v = W[(size_t)n * 1521 + k]; break;
    case 1: v = W[(size_t)n * 2496 + k]; break;
    case 2: { int d = k / MP, m = k - (k / MP) * MP;
              if (m < MM) v = W[(size_t)(m * DD + d) * 256 + n]; } break;
    case 3: v = W[(size_t)k * 128 + n]; break;
    case 4: v = W[(size_t)k * 64 + n]; break;
  }
  frag[gid] = (_Float16)v;
}

// ---------------------------------------------------------------------------
// CIN layer 0: x1[r,o] = relu( sum_{h,m} x0[r,h]*x0[r,m]*w0[o,h*39+m] + b0[o] )
// One block = 16 rows = one batch element b (rows r=b*16+d). 4 waves x 2 ntiles.
// o<64 -> relu + sum over d -> cin0s[b,o]; o>=64 -> store f16 hid[r,o-64].
// ---------------------------------------------------------------------------
__global__ void cin0_kernel(const _Float16* __restrict__ x0t,
                            const _Float16* __restrict__ w0frag,
                            const float* __restrict__ cin_b0,
                            float* __restrict__ cin0s,
                            _Float16* __restrict__ hid) {
  __shared__ __align__(16) _Float16 xs[16 * MP];   // 1280 B = 80 x 16 B
  __shared__ ushort2 hm[1536];
  const int b = blockIdx.x, r0 = b * 16, tid = threadIdx.x;

  // async-stage the 16x40 f16 tile (contiguous in x0t): 80 B128 chunks
  if (tid < 80) {
    unsigned ldsoff = (unsigned)(uintptr_t)(&xs[tid * 8]);
    async_g2l_b128(ldsoff, x0t + (size_t)r0 * MP + tid * 8);
  }
  for (int k = tid; k < 1536; k += 128) {
    int h = k / MM, m = k - h * MM;
    if (k >= 1521) { h = MM; m = MM; }  // map pad to the zero column
    hm[k] = make_ushort2((unsigned short)h, (unsigned short)m);
  }
  wait_asynccnt0();
  __syncthreads();

  const int lane = tid & 31, wave = tid >> 5;
  const _Float16* xr = &xs[(lane & 15) * MP];
  const int khi = (lane >> 4) * 8;
  v8f z = {};
  v8f acc[2] = {z, z};

  for (int kt = 0; kt < 48; ++kt) {
    HU16 au;
#pragma unroll
    for (int j = 0; j < 16; ++j) {
      int kl = khi + ((j < 8) ? j : (j + 8));
      ushort2 t = hm[kt * 32 + kl];
      au.h[j] = xr[t.x] * xr[t.y];
    }
#pragma unroll
    for (int t2 = 0; t2 < 2; ++t2) {
      int nt = wave * 2 + t2;
      const _Float16* bp = w0frag + ((size_t)(kt * 8 + nt) * 32 + lane) * 16;
      v16h bf = cat8(*(const v8h*)bp, *(const v8h*)(bp + 8));
      acc[t2] = wmma_f16(au.v, bf, acc[t2]);
    }
  }

  const int mbase = (lane >> 4) * 8;
#pragma unroll
  for (int t2 = 0; t2 < 2; ++t2) {
    int nt = wave * 2 + t2;
    int n  = nt * 16 + (lane & 15);
    float bias = cin_b0[n];
    if (nt < 4) {  // direct channels: relu then sum over d (the 16 rows of tile)
      float s = 0.0f;
#pragma unroll
      for (int v = 0; v < 8; ++v) s += fmaxf(acc[t2][v] + bias, 0.0f);
      s += __shfl_xor(s, 16);
      if (lane < 16) cin0s[(size_t)b * 64 + n] = s;
    } else {       // hidden channels: relu, store f16 transposed [r][64]
      int c = n - 64;
#pragma unroll
      for (int v = 0; v < 8; ++v) {
        float y = fmaxf(acc[t2][v] + bias, 0.0f);
        hid[(size_t)(r0 + mbase + v) * 64 + c] = (_Float16)y;
      }
    }
  }
}

// ---------------------------------------------------------------------------
// CIN layer 1: x2[r,o] = relu( sum_{h<64,m<39} hid[r,h]*x0[r,m]*w1[o,h*39+m]+b )
// then sum over d -> x2s[b, 0..127].  K = 2496 = 78 chunks of 32.
// ---------------------------------------------------------------------------
__global__ void cin1_kernel(const _Float16* __restrict__ x0t,
                            const _Float16* __restrict__ hid,
                            const _Float16* __restrict__ w1frag,
                            const float* __restrict__ cin_b1,
                            float* __restrict__ x2s) {
  // per row (208 B pitch): [0..63]=hid, [64..103]=x0(pad)
  __shared__ __align__(16) _Float16 xs[16 * 104];
  __shared__ ushort2 hm[2496];
  const int b = blockIdx.x, r0 = b * 16, tid = threadIdx.x;

  // async-stage hid tile: 16 rows x 64 halves = 128 B128 chunks (one per thread)
  {
    int row = tid >> 3, c8 = (tid & 7) * 8;
    unsigned ldsoff = (unsigned)(uintptr_t)(&xs[row * 104 + c8]);
    async_g2l_b128(ldsoff, hid + (size_t)(r0 + row) * 64 + c8);
  }
  // async-stage x0 tile: 16 rows x 40 halves = 80 B128 chunks
  if (tid < 80) {
    int row = tid / 5, m8 = (tid % 5) * 8;
    unsigned ldsoff = (unsigned)(uintptr_t)(&xs[row * 104 + 64 + m8]);
    async_g2l_b128(ldsoff, x0t + (size_t)(r0 + row) * MP + m8);
  }
  for (int k = tid; k < 2496; k += 128) {
    int h = k / MM, m = k - h * MM;
    hm[k] = make_ushort2((unsigned short)h, (unsigned short)(64 + m));
  }
  wait_asynccnt0();
  __syncthreads();

  const int lane = tid & 31, wave = tid >> 5;
  const _Float16* xr = &xs[(lane & 15) * 104];
  const int khi = (lane >> 4) * 8;
  v8f z = {};
  v8f acc[2] = {z, z};

  for (int kt = 0; kt < 78; ++kt) {
    HU16 au;
#pragma unroll
    for (int j = 0; j < 16; ++j) {
      int kl = khi + ((j < 8) ? j : (j + 8));
      ushort2 t = hm[kt * 32 + kl];
      au.h[j] = xr[t.x] * xr[t.y];
    }
#pragma unroll
    for (int t2 = 0; t2 < 2; ++t2) {
      int nt = wave * 2 + t2;
      const _Float16* bp = w1frag + ((size_t)(kt * 8 + nt) * 32 + lane) * 16;
      v16h bf = cat8(*(const v8h*)bp, *(const v8h*)(bp + 8));
      acc[t2] = wmma_f16(au.v, bf, acc[t2]);
    }
  }

#pragma unroll
  for (int t2 = 0; t2 < 2; ++t2) {
    int nt = wave * 2 + t2;
    int n  = nt * 16 + (lane & 15);
    float bias = cin_b1[n];
    float s = 0.0f;
#pragma unroll
    for (int v = 0; v < 8; ++v) s += fmaxf(acc[t2][v] + bias, 0.0f);
    s += __shfl_xor(s, 16);
    if (lane < 16) x2s[(size_t)b * 128 + n] = s;
  }
}

// ---------------------------------------------------------------------------
// Generic dense GEMM: Y[8192,N] = A[8192,Kpad](f16) @ Bfrag + bias
// Also accumulates per-column sum/sumsq (for batch-norm) into stats[2*N].
// Block = 4 waves, 16 rows; wave handles NT/4 n-tiles. Grid = 512 blocks.
// ---------------------------------------------------------------------------
__global__ void dense_gemm_kernel(const _Float16* __restrict__ A,
                                  const _Float16* __restrict__ Bfrag,
                                  const float* __restrict__ bias,
                                  float* __restrict__ Y,
                                  float* __restrict__ stats,
                                  int Kpad, int KC, int NT, int N) {
  const int tid = threadIdx.x, lane = tid & 31, wave = tid >> 5;
  const int r0 = blockIdx.x * 16;
  const int khi = (lane >= 16) ? 8 : 0;
  const _Float16* Ar = A + (size_t)(r0 + (lane & 15)) * Kpad + khi;
  const int tpw = NT >> 2;   // n-tiles per wave: 4, 2, or 1
  v8f z = {};
  v8f acc[4] = {z, z, z, z};

  for (int kt = 0; kt < KC; ++kt) {
    v8h a0 = *(const v8h*)(Ar + kt * 32);
    v8h a1 = *(const v8h*)(Ar + kt * 32 + 16);
    v16h a = cat8(a0, a1);
#pragma unroll
    for (int t2 = 0; t2 < 4; ++t2) {
      if (t2 >= tpw) break;
      int nt = wave * tpw + t2;
      const _Float16* bp = Bfrag + ((size_t)(kt * NT + nt) * 32 + lane) * 16;
      v16h bf = cat8(*(const v8h*)bp, *(const v8h*)(bp + 8));
      acc[t2] = wmma_f16(a, bf, acc[t2]);
    }
  }

  const int mbase = (lane >= 16) ? 8 : 0;
#pragma unroll
  for (int t2 = 0; t2 < 4; ++t2) {
    if (t2 >= tpw) break;
    int nt = wave * tpw + t2;
    int n  = nt * 16 + (lane & 15);
    float bs = bias[n];
    float s = 0.0f, s2 = 0.0f;
#pragma unroll
    for (int v = 0; v < 8; ++v) {
      float y = acc[t2][v] + bs;
      Y[(size_t)(r0 + mbase + v) * N + n] = y;
      s += y; s2 += y * y;
    }
    s  += __shfl_xor(s, 16);
    s2 += __shfl_xor(s2, 16);
    if (lane < 16) {
      atomicAdd(&stats[2 * n],     s);
      atomicAdd(&stats[2 * n + 1], s2);
    }
  }
}

// ---------------------------------------------------------------------------
// BN (batch stats) + relu; emit f16 (next GEMM input) and/or f32.
// ---------------------------------------------------------------------------
__global__ void bn_relu_kernel(const float* __restrict__ Y,
                               const float* __restrict__ stats,
                               const float* __restrict__ g,
                               const float* __restrict__ be,
                               _Float16* __restrict__ out16,
                               float* __restrict__ out32,
                               int N, int total) {
  int idx = blockIdx.x * blockDim.x + threadIdx.x;
  if (idx >= total) return;
  int n = idx % N;
  const float inv_b = 1.0f / (float)B_SZ;
  float mu  = stats[2 * n] * inv_b;
  float var = stats[2 * n + 1] * inv_b - mu * mu;
  float y = (Y[idx] - mu) * rsqrtf(var + 1e-5f) * g[n] + be[n];
  y = fmaxf(y, 0.0f);
  if (out16) out16[idx] = (_Float16)y;
  if (out32) out32[idx] = y;
}

// ---------------------------------------------------------------------------
// Final: logit = fw . [lin(1), cin0s(64), x2s(128), h2(64)]; sigmoid.
// ---------------------------------------------------------------------------
__global__ void final_kernel(const float* __restrict__ linS,
                             const float* __restrict__ cin0s,
                             const float* __restrict__ x2s,
                             const float* __restrict__ h2,
                             const float* __restrict__ fw,
                             float* __restrict__ out) {
  int b = blockIdx.x * blockDim.x + threadIdx.x;
  if (b >= B_SZ) return;
  float acc = fw[0] * (*linS);
#pragma unroll 4
  for (int o = 0; o < 64; ++o)  acc += fw[1 + o]   * cin0s[(size_t)b * 64 + o];
#pragma unroll 4
  for (int o = 0; o < 128; ++o) acc += fw[65 + o]  * x2s[(size_t)b * 128 + o];
#pragma unroll 4
  for (int j = 0; j < 64; ++j)  acc += fw[193 + j] * h2[(size_t)b * 64 + j];
  out[b] = 1.0f / (1.0f + __expf(-acc));
}

// ---------------------------------------------------------------------------
extern "C" void kernel_launch(void* const* d_in, const int* in_sizes, int n_in,
                              void* d_out, int out_size, void* d_ws, size_t ws_size,
                              hipStream_t stream) {
  (void)in_sizes; (void)n_in; (void)out_size; (void)ws_size;
  const float* numeric  = (const float*)d_in[0];
  const int*   cat_idx  = (const int*)  d_in[1];
  const float* num_w    = (const float*)d_in[2];
  const float* num_b    = (const float*)d_in[3];
  const float* cat_emb  = (const float*)d_in[4];
  const float* lin_tab  = (const float*)d_in[5];
  const float* lin_bias = (const float*)d_in[6];
  const float* cin_w0   = (const float*)d_in[7];
  const float* cin_b0   = (const float*)d_in[8];
  const float* cin_w1   = (const float*)d_in[9];
  const float* cin_b1   = (const float*)d_in[10];
  const float* dw0 = (const float*)d_in[11];
  const float* db0 = (const float*)d_in[12];
  const float* dg0 = (const float*)d_in[13];
  const float* dbe0= (const float*)d_in[14];
  const float* dw1 = (const float*)d_in[15];
  const float* db1 = (const float*)d_in[16];
  const float* dg1 = (const float*)d_in[17];
  const float* dbe1= (const float*)d_in[18];
  const float* dw2 = (const float*)d_in[19];
  const float* db2 = (const float*)d_in[20];
  const float* dg2 = (const float*)d_in[21];
  const float* dbe2= (const float*)d_in[22];
  const float* fw  = (const float*)d_in[23];

  char* ws = (char*)d_ws;
  size_t off = 0;
  auto alloc = [&](size_t bytes) -> void* {
    void* p = ws + off;
    off = (off + bytes + 255) & ~(size_t)255;
    return p;
  };
  _Float16* x0t   = (_Float16*)alloc((size_t)B_SZ * DD * MP * 2);   // 10.5 MB
  _Float16* hid   = (_Float16*)alloc((size_t)B_SZ * DD * 64 * 2);   // 16.8 MB
  float*    cin0s = (float*)   alloc((size_t)B_SZ * 64 * 4);
  float*    x2s   = (float*)   alloc((size_t)B_SZ * 128 * 4);
  _Float16* f0    = (_Float16*)alloc((size_t)48 * 8  * 512 * 2);
  _Float16* f1    = (_Float16*)alloc((size_t)78 * 8  * 512 * 2);
  _Float16* fd0   = (_Float16*)alloc((size_t)20 * 16 * 512 * 2);
  _Float16* fd1   = (_Float16*)alloc((size_t)8  * 8  * 512 * 2);
  _Float16* fd2   = (_Float16*)alloc((size_t)4  * 4  * 512 * 2);
  float*    y0    = (float*)   alloc((size_t)B_SZ * 256 * 4);
  float*    y1    = (float*)   alloc((size_t)B_SZ * 128 * 4);
  float*    y2    = (float*)   alloc((size_t)B_SZ * 64 * 4);
  _Float16* h0f   = (_Float16*)alloc((size_t)B_SZ * 256 * 2);
  _Float16* h1f   = (_Float16*)alloc((size_t)B_SZ * 128 * 2);
  float*    h2    = (float*)   alloc((size_t)B_SZ * 64 * 4);
  float*    stats = (float*)   alloc((size_t)(256 + 128 + 64) * 2 * 4);
  float*    linS  = (float*)   alloc(4);

  // 1) zero BN stats (atomically accumulated each call)
  zero_kernel<<<4, 256, 0, stream>>>(stats, 896);
  // 2) embeddings + numeric -> x0 transposed f16 [b][d][m(40)]
  build_x0_kernel<<<(B_SZ * MM + 255) / 256, 256, 0, stream>>>(
      numeric, cat_idx, num_w, num_b, cat_emb, x0t);
  // 3) linear scalar
  lin_kernel<<<1, 64, 0, stream>>>(lin_tab, lin_bias, linS);
  // 4) pre-swizzle all weight matrices into WMMA B-fragment layout
  prep_bfrag_kernel<<<(48*8*512  + 255)/256, 256, 0, stream>>>(cin_w0, f0, 0, 8,  48*8*512);
  prep_bfrag_kernel<<<(78*8*512  + 255)/256, 256, 0, stream>>>(cin_w1, f1, 1, 8,  78*8*512);
  prep_bfrag_kernel<<<(20*16*512 + 255)/256, 256, 0, stream>>>(dw0,   fd0, 2, 16, 20*16*512);
  prep_bfrag_kernel<<<(8*8*512   + 255)/256, 256, 0, stream>>>(dw1,   fd1, 3, 8,  8*8*512);
  prep_bfrag_kernel<<<(4*4*512   + 255)/256, 256, 0, stream>>>(dw2,   fd2, 4, 4,  4*4*512);
  // 5) CIN layers (WMMA, implicit A = outer products built on the fly)
  cin0_kernel<<<B_SZ, 128, 0, stream>>>(x0t, f0, cin_b0, cin0s, hid);
  cin1_kernel<<<B_SZ, 128, 0, stream>>>(x0t, hid, f1, cin_b1, x2s);
  // 6) DNN: GEMM(WMMA) -> BN(batch stats) + relu, x3
  dense_gemm_kernel<<<512, 128, 0, stream>>>(x0t, fd0, db0, y0, stats,       640, 20, 16, 256);
  bn_relu_kernel<<<(B_SZ*256 + 255)/256, 256, 0, stream>>>(y0, stats,       dg0, dbe0, h0f, nullptr, 256, B_SZ*256);
  dense_gemm_kernel<<<512, 128, 0, stream>>>(h0f, fd1, db1, y1, stats + 512, 256, 8,  8,  128);
  bn_relu_kernel<<<(B_SZ*128 + 255)/256, 256, 0, stream>>>(y1, stats + 512, dg1, dbe1, h1f, nullptr, 128, B_SZ*128);
  dense_gemm_kernel<<<512, 128, 0, stream>>>(h1f, fd2, db2, y2, stats + 768, 128, 4,  4,  64);
  bn_relu_kernel<<<(B_SZ*64  + 255)/256, 256, 0, stream>>>(y2, stats + 768, dg2, dbe2, nullptr, h2, 64, B_SZ*64);
  // 7) final concat + dot + sigmoid
  final_kernel<<<B_SZ / 256, 256, 0, stream>>>(linS, cin0s, x2s, h2, fw, (float*)d_out);
}
